// PositionLinearAttention_17944373362865
// MI455X (gfx1250) — matrix operand order repaired and checked by reading, not compiled, not run on hardware
//
#include <hip/hip_runtime.h>
#include <hip/hip_bf16.h>

// ---------------------------------------------------------------------------
// PositionLinearAttention for MI455X (gfx1250), wave32 + WMMA f16 (f32 accum)
// B=8, C=256, Cqk=32, H=W=128, N=16384
// ---------------------------------------------------------------------------

typedef _Float16 h16;
typedef __attribute__((ext_vector_type(16))) _Float16 v16h;
typedef __attribute__((ext_vector_type(8)))  float    v8f;

#define NPOS   16384
#define CCH    256
#define CQK    32
#define NBATCH 8
#define EPSF   1e-6f

union FragH {
    v16h v;
    uint4 u[2];
    h16  h[16];
};

__device__ __forceinline__ v8f wmma_f16(v16h a, v16h b, v8f c) {
    // (neg_a, A, neg_b, B, c_mod, C, reuse_a, reuse_b)
    return __builtin_amdgcn_wmma_f32_16x16x32_f16(false, a, false, b, (short)0, c,
                                                  false, false);
}

// A fragment (16x32 f16) from a row-major [*, 256] f16 matrix.
// Lane L: row = base_row + (L&15); halfs e: k = ks*32 + 16*(e>>3) + 8*(L>>4) + (e&7)
__device__ __forceinline__ v16h load_a_row(const h16* __restrict__ wrow, int ks, int hlf) {
    FragH a;
    const h16* p = wrow + ks * 32 + hlf * 8;
    a.u[0] = *(const uint4*)(p);
    a.u[1] = *(const uint4*)(p + 16);
    return a.v;
}

__device__ __forceinline__ float wave16_sum(float v) {
    v += __shfl_xor(v, 1, 32);
    v += __shfl_xor(v, 2, 32);
    v += __shfl_xor(v, 4, 32);
    v += __shfl_xor(v, 8, 32);
    return v;
}

// ---------------------------------------------------------------------------
// Kernel 1: pack weights (f16), biases (f32), zero accumulators
// grid = 320 x 256
// ---------------------------------------------------------------------------
__global__ __launch_bounds__(256) void pla_prep_kernel(
    const float* __restrict__ Wq, const float* __restrict__ bq,
    const float* __restrict__ Wk, const float* __restrict__ bk,
    const float* __restrict__ Wv, const float* __restrict__ bv,
    h16* __restrict__ Wh, float* __restrict__ biasf,
    float* __restrict__ matrix, float* __restrict__ k_sum,
    float* __restrict__ value_sum)
{
    const int idx = blockIdx.x * 256 + threadIdx.x;      // 0 .. 81919
    if (idx < 320 * 256) {
        const int row = idx >> 8, k = idx & 255;
        float v;
        if (row < 32)      v = Wq[row * 256 + k];
        else if (row < 64) v = Wk[(row - 32) * 256 + k];
        else               v = Wv[(row - 64) * 256 + k];
        Wh[idx] = (h16)v;
    }
    if (idx < 320)
        biasf[idx] = (idx < 32) ? bq[idx] : ((idx < 64) ? bk[idx - 32] : bv[idx - 64]);
    if (idx < NBATCH * CQK * CCH) matrix[idx]    = 0.f;
    if (idx < NBATCH * CQK)       k_sum[idx]     = 0.f;
    if (idx < NBATCH * CCH)       value_sum[idx] = 0.f;
}

// ---------------------------------------------------------------------------
// Kernel 2: QKV projection GEMM (WMMA) + l2-norm + reductions + layout
//  grid = B*128 blocks of 256 threads (8 waves); each block: one batch b,
//  one 128-column chunk of N.  Wave w owns column-tile w (16 columns).
//  LDS (dynamic, 78336 B):
//    phase 1: x tile f16, layout [n(128)][c(256)], stride 272 halfs
//    phase 2: output staging [row(288)][col(128)], stride 136 halfs
//             rows 0..255 = V, rows 256..287 = Kn
// ---------------------------------------------------------------------------
__global__ __launch_bounds__(256) void pla_qkv_kernel(
    const float* __restrict__ x, const h16* __restrict__ Wh,
    const float* __restrict__ biasf,
    h16* __restrict__ QnT, h16* __restrict__ KnH, h16* __restrict__ Vh,
    float* __restrict__ k_sum, float* __restrict__ value_sum)
{
    extern __shared__ h16 sm[];
    const int tid = threadIdx.x;
    const int b     = blockIdx.x >> 7;
    const int chunk = blockIdx.x & 127;
    const int n0    = chunk * 128;
    const int w      = tid >> 5;
    const int L      = tid & 31;
    const int lane15 = L & 15;
    const int hlf    = L >> 4;

    // L2 warm-up hint for the (hot, shared) weight matrix
    for (int i = tid; i < 320; i += 256)
        __builtin_prefetch(Wh + (size_t)i * 256, 0, 3);

    // ---- phase 1: stage x chunk as f16, LDS layout [n][c] stride 272 ----
    {
        const int n4 = (tid & 31) * 4;
        const int cw = tid >> 5;
        #pragma unroll 4
        for (int it = 0; it < 32; ++it) {
            const int c = it * 8 + cw;
            const float4 f =
                *(const float4*)(x + ((size_t)(b * CCH + c)) * NPOS + n0 + n4);
            sm[(size_t)(n4 + 0) * 272 + c] = (h16)f.x;
            sm[(size_t)(n4 + 1) * 272 + c] = (h16)f.y;
            sm[(size_t)(n4 + 2) * 272 + c] = (h16)f.z;
            sm[(size_t)(n4 + 3) * 272 + c] = (h16)f.w;
        }
    }
    __syncthreads();

    // ---- phase 2: per-wave B fragments (x columns of this col-tile) ----
    // B frag (32x16): lane L -> column n = w*16 + (L&15); halfs e: k = ks*32 + 16*(e>>3) + 8*hlf + (e&7)
    const int nl = w * 16 + lane15;
    FragH bf[8];
    #pragma unroll
    for (int ks = 0; ks < 8; ++ks) {
        const h16* p = sm + (size_t)nl * 272 + ks * 32 + hlf * 8;
        bf[ks].u[0] = *(const uint4*)(p);
        bf[ks].u[1] = *(const uint4*)(p + 16);
    }
    __syncthreads();      // LDS is reused as output staging from here on

    const int n_glob = n0 + w * 16 + lane15;
    const int colL   = w * 16 + lane15;

    // ---- Q: row-tiles 0,1 (channels 0..31), normalize, store QnT[n][32] ----
    {
        v8f q0 = {}, q1 = {};
        const h16* w0 = Wh + (size_t)(0  + lane15) * 256;
        const h16* w1 = Wh + (size_t)(16 + lane15) * 256;
        #pragma unroll
        for (int ks = 0; ks < 8; ++ks) {
            q0 = wmma_f16(load_a_row(w0, ks, hlf), bf[ks].v, q0);
            q1 = wmma_f16(load_a_row(w1, ks, hlf), bf[ks].v, q1);
        }
        float c0[8], c1[8];
        float ssq = 0.f;
        #pragma unroll
        for (int r = 0; r < 8; ++r) {
            c0[r] = q0[r] + biasf[hlf * 8 + r];
            c1[r] = q1[r] + biasf[16 + hlf * 8 + r];
            ssq += c0[r] * c0[r] + c1[r] * c1[r];
        }
        ssq += __shfl_xor(ssq, 16, 32);      // combine the two row halves of column n
        const float rn = rsqrtf(ssq);
        union { uint4 u; h16 h[8]; } p0, p1;
        #pragma unroll
        for (int r = 0; r < 8; ++r) {
            p0.h[r] = (h16)(c0[r] * rn);
            p1.h[r] = (h16)(c1[r] * rn);
        }
        h16* qdst = QnT + ((size_t)b * NPOS + n_glob) * 32;
        *(uint4*)(qdst + hlf * 8)      = p0.u;    // rows 8*hlf .. 8*hlf+7
        *(uint4*)(qdst + 16 + hlf * 8) = p1.u;    // rows 16+8*hlf ..
    }

    // ---- K: row-tiles 2,3 (channels 32..63), normalize, k_sum, stage Kn ----
    {
        v8f k0 = {}, k1 = {};
        const h16* w0 = Wh + (size_t)(32 + lane15) * 256;
        const h16* w1 = Wh + (size_t)(48 + lane15) * 256;
        #pragma unroll
        for (int ks = 0; ks < 8; ++ks) {
            k0 = wmma_f16(load_a_row(w0, ks, hlf), bf[ks].v, k0);
            k1 = wmma_f16(load_a_row(w1, ks, hlf), bf[ks].v, k1);
        }
        float c0[8], c1[8];
        float ssq = 0.f;
        #pragma unroll
        for (int r = 0; r < 8; ++r) {
            c0[r] = k0[r] + biasf[32 + hlf * 8 + r];
            c1[r] = k1[r] + biasf[48 + hlf * 8 + r];
            ssq += c0[r] * c0[r] + c1[r] * c1[r];
        }
        ssq += __shfl_xor(ssq, 16, 32);
        const float rn = rsqrtf(ssq);
        #pragma unroll
        for (int r = 0; r < 8; ++r) {
            c0[r] *= rn;
            c1[r] *= rn;
            sm[(size_t)(256 + hlf * 8 + r)      * 136 + colL] = (h16)c0[r];
            sm[(size_t)(256 + 16 + hlf * 8 + r) * 136 + colL] = (h16)c1[r];
        }
        #pragma unroll
        for (int r = 0; r < 8; ++r) {
            const float s0 = wave16_sum(c0[r]);
            const float s1 = wave16_sum(c1[r]);
            if (lane15 == 0) {
                atomicAdd(&k_sum[b * CQK + hlf * 8 + r],      s0);
                atomicAdd(&k_sum[b * CQK + 16 + hlf * 8 + r], s1);
            }
        }
    }

    // ---- V: row-tiles 4..19 (channels 0..255), value_sum, stage V ----
    for (int rt = 0; rt < 16; ++rt) {
        v8f acc = {};
        const h16* wr = Wh + (size_t)((4 + rt) * 16 + lane15) * 256;
        #pragma unroll
        for (int ks = 0; ks < 8; ++ks)
            acc = wmma_f16(load_a_row(wr, ks, hlf), bf[ks].v, acc);
        float vv[8];
        #pragma unroll
        for (int r = 0; r < 8; ++r) {
            vv[r] = acc[r] + biasf[64 + rt * 16 + hlf * 8 + r];
            sm[(size_t)(rt * 16 + hlf * 8 + r) * 136 + colL] = (h16)vv[r];
        }
        #pragma unroll
        for (int r = 0; r < 8; ++r) {
            const float s = wave16_sum(vv[r]);
            if (lane15 == 0)
                atomicAdd(&value_sum[b * CCH + rt * 16 + hlf * 8 + r], s);
        }
    }
    __syncthreads();

    // ---- cooperative, coalesced f16 store of staged V (256 rows) + Kn (32) ----
    {
        const int col8  = tid & 15;         // 16 threads x 8 halfs = 128 cols
        const int rbase = tid >> 4;
        #pragma unroll 2
        for (int i = 0; i < 18; ++i) {
            const int row = i * 16 + rbase;                  // 0..287
            const uint4 v = *(const uint4*)(sm + (size_t)row * 136 + col8 * 8);
            h16* dst;
            if (row < 256)
                dst = Vh  + ((size_t)(b * CCH + row)) * NPOS + n0 + col8 * 8;
            else
                dst = KnH + ((size_t)(b * CQK + (row - 256))) * NPOS + n0 + col8 * 8;
            *(uint4*)dst = v;
        }
    }
}

// ---------------------------------------------------------------------------
// Kernel 3: matrix[b][32][256] += Kn[32xN] * V^T (K-dim = N), WMMA + f32 atomics
//  grid = B*64 blocks; block = (batch, 256-position slab = 8 k-steps)
//  8 waves x 4 output tiles = 32 tiles (2 m-tiles x 16 c-tiles)
// ---------------------------------------------------------------------------
__global__ __launch_bounds__(256) void pla_matrix_kernel(
    const h16* __restrict__ KnH, const h16* __restrict__ Vh,
    float* __restrict__ matrix)
{
    const int tid  = threadIdx.x;
    const int b    = blockIdx.x >> 6;
    const int slab = blockIdx.x & 63;
    const int nb0  = slab * 256;
    const int w = tid >> 5, L = tid & 31, lane15 = L & 15, hlf = L >> 4;

    #pragma unroll
    for (int t = 0; t < 4; ++t) {
        const int T  = w * 4 + t;
        const int mt = T & 1;
        const int ct = T >> 1;
        const h16* arow = KnH + ((size_t)(b * CQK + mt * 16 + lane15)) * NPOS;
        const h16* brow = Vh  + ((size_t)(b * CCH + ct * 16 + lane15)) * NPOS;
        v8f acc = {};
        #pragma unroll
        for (int ks = 0; ks < 8; ++ks) {
            const int nb = nb0 + ks * 32;
            FragH a, bb;
            a.u[0]  = *(const uint4*)(arow + nb + hlf * 8);
            a.u[1]  = *(const uint4*)(arow + nb + 16 + hlf * 8);
            const h16* bp = brow + nb + hlf * 8;
            bb.u[0] = *(const uint4*)(bp);
            bb.u[1] = *(const uint4*)(bp + 16);
            acc = wmma_f16(a.v, bb.v, acc);
        }
        #pragma unroll
        for (int r = 0; r < 8; ++r)
            atomicAdd(&matrix[((size_t)(b * CQK + mt * 16 + hlf * 8 + r)) * CCH +
                              ct * 16 + lane15],
                      acc[r]);
    }
}

// ---------------------------------------------------------------------------
// Kernel 4: matrixT[b][256][32] (f16) = matrix[b][32][256] (f32)
//  grid = 256 x 256 (65536 elements)
// ---------------------------------------------------------------------------
__global__ __launch_bounds__(256) void pla_matT_kernel(
    const float* __restrict__ matrix, h16* __restrict__ matT)
{
    const int idx = blockIdx.x * 256 + threadIdx.x;      // (b, c, m) order
    const int b = idx >> 13;
    const int c = (idx >> 5) & 255;
    const int m = idx & 31;
    matT[idx] = (h16)matrix[((size_t)(b * CQK + m)) * CCH + c];
}

// ---------------------------------------------------------------------------
// Kernel 5: out = x + gamma * tailor(n) * (value_sum + matrixT @ Qn[:,n])
//  grid = B*128 blocks; block = (batch, 128-column chunk); wave w = col-tile w
//  K-dim = 32 -> a single 16x16x32 WMMA per output tile.
// ---------------------------------------------------------------------------
__global__ __launch_bounds__(256) void pla_out_kernel(
    const float* __restrict__ x, const h16* __restrict__ QnT,
    const h16* __restrict__ matT, const float* __restrict__ k_sum,
    const float* __restrict__ value_sum, const float* __restrict__ gamma,
    float* __restrict__ out)
{
    const int tid   = threadIdx.x;
    const int b     = blockIdx.x >> 7;
    const int chunk = blockIdx.x & 127;
    const int n0    = chunk * 128;
    const int w = tid >> 5, L = tid & 31, lane15 = L & 15, hlf = L >> 4;
    const int n_glob = n0 + w * 16 + lane15;

    // B fragment: column n of Qn (32 contiguous f16), halfs e: k = 16*hlf + e
    FragH bq;
    const h16* qp = QnT + ((size_t)b * NPOS + n_glob) * 32 + hlf * 16;
    bq.u[0] = *(const uint4*)(qp);
    bq.u[1] = *(const uint4*)(qp + 8);

    // tailor(n) = 1 / (N + Qn[:,n] . (k_sum + eps))
    float part = 0.f;
    const float* ksp = k_sum + b * CQK + hlf * 16;
    #pragma unroll
    for (int j = 0; j < 16; ++j)
        part += (float)bq.h[j] * (ksp[j] + EPSF);
    part += __shfl_xor(part, 16, 32);
    const float tail = 1.0f / ((float)NPOS + part);
    const float g    = gamma[0];

    #pragma unroll 2
    for (int ct = 0; ct < 16; ++ct) {
        FragH a;
        const h16* ap = matT + ((size_t)(b * CCH + ct * 16 + lane15)) * 32 + hlf * 16;
        a.u[0] = *(const uint4*)(ap);
        a.u[1] = *(const uint4*)(ap + 8);
        v8f acc = {};
        acc = wmma_f16(a.v, bq.v, acc);
        #pragma unroll
        for (int r = 0; r < 8; ++r) {
            const int c = ct * 16 + hlf * 8 + r;
            const size_t idx = ((size_t)(b * CCH + c)) * NPOS + n_glob;
            out[idx] = x[idx] + g * tail * (value_sum[b * CCH + c] + acc[r]);
        }
    }
}

// ---------------------------------------------------------------------------
// Workspace layout (bytes):
//   0        Wh       81920 f16      (160 KB)
//   163840   biasf    320 f32
//   165120   k_sum    256 f32
//   166144   value_sum 2048 f32
//   174336   matrix   65536 f32      (256 KB)
//   436480   matrixT  65536 f16      (128 KB)
//   1 MB     QnT      8*16384*32 f16 (8 MB)
//   9 MB     KnH      8*32*16384 f16 (8 MB)
//   17 MB    Vh       8*256*16384 f16 (64 MB)   total ~81 MB
// ---------------------------------------------------------------------------
extern "C" void kernel_launch(void* const* d_in, const int* in_sizes, int n_in,
                              void* d_out, int out_size, void* d_ws, size_t ws_size,
                              hipStream_t stream)
{
    (void)in_sizes; (void)n_in; (void)out_size; (void)ws_size;

    const float* x     = (const float*)d_in[0];
    const float* Wq    = (const float*)d_in[1];
    const float* bq    = (const float*)d_in[2];
    const float* Wk    = (const float*)d_in[3];
    const float* bk    = (const float*)d_in[4];
    const float* Wv    = (const float*)d_in[5];
    const float* bv    = (const float*)d_in[6];
    const float* gamma = (const float*)d_in[7];
    float*       out   = (float*)d_out;

    char* ws = (char*)d_ws;
    h16*   Wh        = (h16*)(ws + 0);
    float* biasf     = (float*)(ws + 163840);
    float* k_sum     = (float*)(ws + 165120);
    float* value_sum = (float*)(ws + 166144);
    float* matrix    = (float*)(ws + 174336);
    h16*   matT      = (h16*)(ws + 436480);
    h16*   QnT       = (h16*)(ws + ((size_t)1  << 20));
    h16*   KnH       = (h16*)(ws + ((size_t)9  << 20));
    h16*   Vh        = (h16*)(ws + ((size_t)17 << 20));

    pla_prep_kernel<<<320, 256, 0, stream>>>(Wq, bq, Wk, bk, Wv, bv,
                                             Wh, biasf, matrix, k_sum, value_sum);

    pla_qkv_kernel<<<NBATCH * 128, 256, 78336, stream>>>(
        x, Wh, biasf, QnT, KnH, Vh, k_sum, value_sum);

    pla_matrix_kernel<<<NBATCH * 64, 256, 0, stream>>>(KnH, Vh, matrix);

    pla_matT_kernel<<<256, 256, 0, stream>>>(matrix, matT);

    pla_out_kernel<<<NBATCH * 128, 256, 0, stream>>>(
        x, QnT, matT, k_sum, value_sum, gamma, out);
}
